// AdaAttentionalPropagation_89739046683470
// MI455X (gfx1250) — matrix operand order repaired and checked by reading, not compiled, not run on hardware
//
#include <hip/hip_runtime.h>
#include <hip/hip_bf16.h>

// ---------------------------------------------------------------------------
// Problem constants (match reference)
// ---------------------------------------------------------------------------
#define BATCH 8
#define DMODEL 256
#define NTOK 2048L
#define NHEAD 4
#define DHEAD 64

typedef __attribute__((ext_vector_type(16))) __bf16 v16bf;
typedef __attribute__((ext_vector_type(8)))  __bf16 bf16x8;
typedef __attribute__((ext_vector_type(8)))  float  v8f;

#define WMMA_BF16(A, B, C) \
  __builtin_amdgcn_wmma_f32_16x16x32_bf16(false, (A), false, (B), (short)0, (C), false, false)

static __device__ __forceinline__ v8f zero_v8f() {
  v8f z = {0.f, 0.f, 0.f, 0.f, 0.f, 0.f, 0.f, 0.f};
  return z;
}

// ---------------------------------------------------------------------------
// CDNA5 async global->LDS copy (ASYNCcnt-tracked, no VGPR round trip).
// VDST holds the per-lane LDS byte address; for a generic pointer to LDS the
// low 32 bits ARE the LDS offset (flat aperture rule, ISA 10.2).
// ---------------------------------------------------------------------------
static __device__ __forceinline__ void async_load_b128_to_lds(const void* gptr, void* lptr) {
  const unsigned lds_off = (unsigned)(unsigned long long)lptr;
  asm volatile("global_load_async_to_lds_b128 %0, %1, off"
               :: "v"(lds_off), "v"((unsigned long long)gptr)
               : "memory");
}

static __device__ __forceinline__ void wait_asynccnt0() {
#if __has_builtin(__builtin_amdgcn_s_wait_asynccnt)
  __builtin_amdgcn_s_wait_asynccnt(0);
#else
  asm volatile("s_wait_asynccnt 0x0" ::: "memory");
#endif
}

// ---------------------------------------------------------------------------
// K0: 32x32 tile transpose + fp32->bf16. src is channel-major [B][256][N],
// dst is token-major [B][N][dstStride] (columns 0..255 filled).
// ---------------------------------------------------------------------------
__global__ __launch_bounds__(256) void transpose_cvt_kernel(
    const float* __restrict__ src, __bf16* __restrict__ dst, int dstStride) {
  __shared__ float tile[32][33];
  const int b = blockIdx.z;
  const int c0 = blockIdx.y * 32, n0 = blockIdx.x * 32;
  const int tcol = threadIdx.x & 31, trow = threadIdx.x >> 5;  // 32 x 8
  const float* s = src + ((long)b * DMODEL + c0) * NTOK + n0;
#pragma unroll
  for (int p = 0; p < 4; ++p)
    tile[p * 8 + trow][tcol] = s[(long)(p * 8 + trow) * NTOK + tcol];
  __syncthreads();
  __bf16* d = dst + ((long)b * NTOK + n0) * dstStride + c0;
#pragma unroll
  for (int p = 0; p < 4; ++p)
    d[(long)(p * 8 + trow) * dstStride + tcol] = (__bf16)tile[tcol][p * 8 + trow];
}

// ---------------------------------------------------------------------------
// K1: pointwise-conv GEMM  out[oc, n] = sum_k W[oc,k] * X[n,k] + bias[oc]
//   X is token-major -> B fragments are two global_load_b128 per lane.
//   Workgroup = 128 threads (4 waves); one 16-row W block staged in LDS.
// MODE 0: bf16 out, token-major [n][ocTotal] via LDS-transpose epilogue
// MODE 1: bf16 out, head layout [h][n][dh]  (c = dh*H + h)
// MODE 3: f32  out, channel-major [oc][N]
// ---------------------------------------------------------------------------
template <int IC, int MODE>
__global__ __launch_bounds__(128) void gemm_bias_kernel(
    const float* __restrict__ W, const float* __restrict__ bias,
    const __bf16* __restrict__ X, long xStride, int xRowStride,
    void* __restrict__ outp, long oStride, int ocTotal) {
  constexpr int ICP = IC + 8;  // 16B-aligned rows, conflict-free A reads
  __shared__ __align__(16) __bf16 ldsW[16 * ICP];

  const int b    = blockIdx.z;
  const int oc0  = blockIdx.y * 16;
  const int n0   = blockIdx.x * 64;
  const int tid  = threadIdx.x;
  const int wave = tid >> 5, lane = tid & 31;
  const int hi   = lane >> 4, col = lane & 15;

  // stage W row-block (16 x IC) as bf16
  for (int i = tid; i < 16 * IC; i += 128) {
    int r = i / IC, c = i - r * IC;
    ldsW[r * ICP + c] = (__bf16)W[(long)(oc0 + r) * IC + c];
  }
  __syncthreads();

  v8f acc = zero_v8f();
  const int nw = n0 + wave * 16 + col;
  const __bf16* xrow = X + (long)b * xStride + (long)nw * xRowStride;

#pragma unroll
  for (int kc = 0; kc < IC / 32; ++kc) {
    // A fragment: A[m=col][k], k = kc*32 + 16*(j>>3) + 8*hi + (j&7)
    const __bf16* arow = &ldsW[col * ICP + kc * 32];
    bf16x8 a0 = *(const bf16x8*)(arow + 8 * hi);
    bf16x8 a1 = *(const bf16x8*)(arow + 16 + 8 * hi);
    v16bf a;
#pragma unroll
    for (int j = 0; j < 8; ++j) { a[j] = a0[j]; a[8 + j] = a1[j]; }

    // B fragment: B[k][col], k = kc*32 + 16*(j>>3) + 8*hi + (j&7)
    const __bf16* xp = xrow + kc * 32 + 8 * hi;
    bf16x8 b0 = *(const bf16x8*)xp;
    bf16x8 b1 = *(const bf16x8*)(xp + 16);
    v16bf bb;
#pragma unroll
    for (int j = 0; j < 8; ++j) { bb[j] = b0[j]; bb[8 + j] = b1[j]; }

    acc = WMMA_BF16(a, bb, acc);
  }

  float vals[8];
#pragma unroll
  for (int r = 0; r < 8; ++r) vals[r] = acc[r] + bias[oc0 + r + 8 * hi];

  if (MODE == 3) {
    float* o = (float*)outp + (long)b * oStride;
#pragma unroll
    for (int r = 0; r < 8; ++r) o[(long)(oc0 + r + 8 * hi) * NTOK + nw] = vals[r];
  } else if (MODE == 1) {
    __bf16* o = (__bf16*)outp + (long)b * oStride;
#pragma unroll
    for (int r = 0; r < 8; ++r) {
      const int oc = oc0 + r + 8 * hi;
      o[((long)(oc & 3) * NTOK + nw) * DHEAD + (oc >> 2)] = (__bf16)vals[r];
    }
  } else {
    // MODE 0: transpose tile in LDS, store token-major b128 chunks
    __syncthreads();
    __bf16* ldsT = ldsW;  // reuse as [64][24]
#pragma unroll
    for (int r = 0; r < 8; ++r)
      ldsT[(wave * 16 + col) * 24 + r + 8 * hi] = (__bf16)vals[r];
    __syncthreads();
    const int n_l = tid >> 1, half = (tid & 1) * 8;
    __bf16* o = (__bf16*)outp + (long)b * oStride +
                (long)(n0 + n_l) * ocTotal + oc0 + half;
    *(bf16x8*)o = *(const bf16x8*)&ldsT[n_l * 24 + half];
  }
}

// ---------------------------------------------------------------------------
// K2: fused flash attention with multiplicative mask.
//   scores = (q.k)/8 * mask ; online softmax ; O = P.V
//   256 threads = 8 waves; each wave owns 16 query rows.
//   K tile staged via GLOBAL_LOAD_ASYNC_TO_LDS_B128 (ASYNCcnt), V staged
//   transposed so every B fragment is two ds_load_b128.
// q: [B][H][N][DH]  k,v: [B][H][M][DH]  mask: [B][N][M]
// attn out: token-major [B][N][D] with c = dh*H + h
// ---------------------------------------------------------------------------
__global__ __launch_bounds__(256) void flash_attn_kernel(
    const __bf16* __restrict__ qb, const __bf16* __restrict__ kb,
    const __bf16* __restrict__ vb, const float* __restrict__ mask,
    __bf16* __restrict__ attnb) {
  constexpr int DHP = 72;  // K tile row pad (bytes mult of 16, conflict-free)
  constexpr int MP  = 40;  // VT tile row pad
  __shared__ __align__(16) __bf16 ldsK[32 * DHP];     // [m][dh]
  __shared__ __align__(16) __bf16 ldsVT[64 * MP];     // [dh][m]
  __shared__ __align__(16) __bf16 ldsP[8 * 16 * 40];  // per-wave P transpose

  const int b = blockIdx.z, h = blockIdx.y;
  const int tid  = threadIdx.x;
  const int wave = tid >> 5, lane = tid & 31;
  const int hi   = lane >> 4, col = lane & 15;
  const int n0   = blockIdx.x * 128 + wave * 16;
  const long bh  = (long)b * NHEAD + h;

  // --- preload the wave's Q fragments (A side), dh=64 -> two K-chunks of 32
  v16bf qf[2];
  const __bf16* qrow = qb + (bh * NTOK + (n0 + col)) * DHEAD;
#pragma unroll
  for (int kc = 0; kc < 2; ++kc) {
    bf16x8 a0 = *(const bf16x8*)(qrow + kc * 32 + 16 * hi);
    bf16x8 a1 = *(const bf16x8*)(qrow + kc * 32 + 16 * hi + 8);
#pragma unroll
    for (int j = 0; j < 8; ++j) { qf[kc][j] = a0[j]; qf[kc][8 + j] = a1[j]; }
  }

  v8f o0 = zero_v8f(), o1 = zero_v8f(), o2 = zero_v8f(), o3 = zero_v8f();
  float mrow[8], lrow[8];
#pragma unroll
  for (int r = 0; r < 8; ++r) { mrow[r] = -3.0e38f; lrow[r] = 0.f; }

  const __bf16* kbase = kb + bh * NTOK * DHEAD;
  const __bf16* vbase = vb + bh * NTOK * DHEAD;
  const float*  mbase = mask + (long)b * NTOK * NTOK;

  const int srow = tid >> 3, scol = (tid & 7) * 8;  // cooperative staging map

  for (int m0 = 0; m0 < (int)NTOK; m0 += 32) {
    // K: async DMA straight into LDS. V: b128 load, transposed scatter store.
    async_load_b128_to_lds(kbase + (long)(m0 + srow) * DHEAD + scol,
                           &ldsK[srow * DHP + scol]);
    bf16x8 vv = *(const bf16x8*)(vbase + (long)(m0 + srow) * DHEAD + scol);
#pragma unroll
    for (int i = 0; i < 8; ++i) ldsVT[(scol + i) * MP + srow] = vv[i];
    if (m0 + 32 < (int)NTOK)  // global_prefetch_b8 of next V tile
      __builtin_prefetch(vbase + (long)(m0 + 32 + srow) * DHEAD + scol, 0, 0);
    wait_asynccnt0();
    __syncthreads();

    // --- load all four K B-fragments, then issue the four S WMMAs
    v16bf bk[4];
#pragma unroll
    for (int kc = 0; kc < 2; ++kc) {
      const __bf16* k0p = &ldsK[col * DHP + kc * 32 + 16 * hi];
      const __bf16* k1p = &ldsK[(16 + col) * DHP + kc * 32 + 16 * hi];
      bf16x8 x0 = *(const bf16x8*)k0p, x1 = *(const bf16x8*)(k0p + 8);
      bf16x8 y0 = *(const bf16x8*)k1p, y1 = *(const bf16x8*)(k1p + 8);
#pragma unroll
      for (int j = 0; j < 8; ++j) {
        bk[kc][j] = x0[j];     bk[kc][8 + j] = x1[j];
        bk[2 + kc][j] = y0[j]; bk[2 + kc][8 + j] = y1[j];
      }
    }
    v8f s0 = zero_v8f(), s1 = zero_v8f();
    s0 = WMMA_BF16(qf[0], bk[0], s0);
    s0 = WMMA_BF16(qf[1], bk[1], s0);
    s1 = WMMA_BF16(qf[0], bk[2], s1);
    s1 = WMMA_BF16(qf[1], bk[3], s1);

    // --- scale, multiplicative mask, online softmax
    float p0[8], p1[8], tmax[8], rs[8];
#pragma unroll
    for (int r = 0; r < 8; ++r) {
      const float* mr = mbase + (long)(n0 + r + 8 * hi) * NTOK + m0;
      p0[r] = s0[r] * 0.125f * mr[col];
      p1[r] = s1[r] * 0.125f * mr[16 + col];
      tmax[r] = fmaxf(p0[r], p1[r]);
    }
#pragma unroll
    for (int msk = 1; msk < 16; msk <<= 1)
#pragma unroll
      for (int r = 0; r < 8; ++r) tmax[r] = fmaxf(tmax[r], __shfl_xor(tmax[r], msk, 32));
#pragma unroll
    for (int r = 0; r < 8; ++r) {
      const float mn = fmaxf(mrow[r], tmax[r]);
      const float sc = __expf(mrow[r] - mn);
      mrow[r] = mn;
      p0[r] = __expf(p0[r] - mn);
      p1[r] = __expf(p1[r] - mn);
      rs[r] = p0[r] + p1[r];
      lrow[r] *= sc;
      o0[r] *= sc; o1[r] *= sc; o2[r] *= sc; o3[r] *= sc;
    }
#pragma unroll
    for (int msk = 1; msk < 16; msk <<= 1)
#pragma unroll
      for (int r = 0; r < 8; ++r) rs[r] += __shfl_xor(rs[r], msk, 32);
#pragma unroll
    for (int r = 0; r < 8; ++r) lrow[r] += rs[r];

    // --- transpose P (C-layout -> A-layout) through per-wave LDS
    __bf16* pw = &ldsP[wave * 640];
#pragma unroll
    for (int r = 0; r < 8; ++r) {
      pw[(r + 8 * hi) * 40 + col]      = (__bf16)p0[r];
      pw[(r + 8 * hi) * 40 + 16 + col] = (__bf16)p1[r];
    }
    const __bf16* pr = &ldsP[wave * 640 + col * 40];
    bf16x8 pa0 = *(const bf16x8*)(pr + 16 * hi);
    bf16x8 pa1 = *(const bf16x8*)(pr + 16 * hi + 8);
    v16bf pa;
#pragma unroll
    for (int j = 0; j < 8; ++j) { pa[j] = pa0[j]; pa[8 + j] = pa1[j]; }

    // --- load all four V fragments, then issue the four P.V WMMAs
    v16bf bv[4];
#pragma unroll
    for (int t = 0; t < 4; ++t) {
      const __bf16* vp = &ldsVT[(t * 16 + col) * MP + 16 * hi];
      bf16x8 vb0 = *(const bf16x8*)vp;
      bf16x8 vb1 = *(const bf16x8*)(vp + 8);
#pragma unroll
      for (int j = 0; j < 8; ++j) { bv[t][j] = vb0[j]; bv[t][8 + j] = vb1[j]; }
    }
    o0 = WMMA_BF16(pa, bv[0], o0);
    o1 = WMMA_BF16(pa, bv[1], o1);
    o2 = WMMA_BF16(pa, bv[2], o2);
    o3 = WMMA_BF16(pa, bv[3], o3);
    __syncthreads();
  }

  // --- epilogue: normalize, store token-major attn [B][N][D], c = dh*H + h
  float inv[8];
#pragma unroll
  for (int r = 0; r < 8; ++r) inv[r] = 1.0f / lrow[r];
  __bf16* ob = attnb + (long)b * NTOK * DMODEL;
#pragma unroll
  for (int t = 0; t < 4; ++t) {
#pragma unroll
    for (int r = 0; r < 8; ++r) {
      const int c = (t * 16 + col) * NHEAD + h;
      const float v = (t == 0 ? o0[r] : t == 1 ? o1[r] : t == 2 ? o2[r] : o3[r]) * inv[r];
      ob[(long)(n0 + r + 8 * hi) * DMODEL + c] = (__bf16)v;
    }
  }
}

// ---------------------------------------------------------------------------
// K3a: InstanceNorm1d stats per (b, channel) over N: stats = {mean, rsqrt}
// ---------------------------------------------------------------------------
__global__ __launch_bounds__(256) void norm_stats_kernel(const float* __restrict__ y,
                                                         float2* __restrict__ stats) {
  const long row = blockIdx.x;  // b*512 + c
  const float* src = y + row * NTOK;
  float s = 0.f, ss = 0.f;
  for (int i = threadIdx.x; i < (int)NTOK; i += 256) {
    const float v = src[i];
    s += v; ss += v * v;
  }
#pragma unroll
  for (int m = 16; m; m >>= 1) { s += __shfl_xor(s, m, 32); ss += __shfl_xor(ss, m, 32); }
  __shared__ float shs[8], shss[8];
  const int w = threadIdx.x >> 5, ln = threadIdx.x & 31;
  if (ln == 0) { shs[w] = s; shss[w] = ss; }
  __syncthreads();
  if (threadIdx.x == 0) {
    float S = 0.f, SS = 0.f;
#pragma unroll
    for (int k = 0; k < 8; ++k) { S += shs[k]; SS += shss[k]; }
    const float mean = S * (1.0f / NTOK);
    const float var  = SS * (1.0f / NTOK) - mean * mean;
    stats[row] = make_float2(mean, rsqrtf(var + 1e-5f));
  }
}

// ---------------------------------------------------------------------------
// K3b: apply norm + ReLU, transpose channel-major f32 -> token-major bf16
// (32-token tiles staged in LDS so both sides stay coalesced b128)
// ---------------------------------------------------------------------------
__global__ __launch_bounds__(256) void norm_apply_kernel(
    const float* __restrict__ y, const float2* __restrict__ stats,
    __bf16* __restrict__ zt) {
  constexpr int CP = 520;  // 512 + 8 pad, rows 16B aligned
  __shared__ __align__(16) __bf16 tile[32 * CP];
  const int b  = blockIdx.y;
  const int n0 = blockIdx.x * 32;
  const int nn = threadIdx.x & 31, cc = threadIdx.x >> 5;
  const float*  yb = y + (long)b * 512 * NTOK;
  const float2* st = stats + (long)b * 512;
  for (int c = cc; c < 512; c += 8) {
    const float2 s = st[c];
    const float v = (yb[(long)c * NTOK + n0 + nn] - s.x) * s.y;
    tile[nn * CP + c] = (__bf16)fmaxf(v, 0.f);
  }
  __syncthreads();
  const int n_l = threadIdx.x >> 3, ch = (threadIdx.x & 7) * 64;
  __bf16* dst = zt + ((long)b * NTOK + n0 + n_l) * 512 + ch;
  const __bf16* srcT = &tile[n_l * CP + ch];
#pragma unroll
  for (int j = 0; j < 8; ++j)
    *(bf16x8*)(dst + j * 8) = *(const bf16x8*)(srcT + j * 8);
}

// ---------------------------------------------------------------------------
// Host orchestration
// ---------------------------------------------------------------------------
extern "C" void kernel_launch(void* const* d_in, const int* in_sizes, int n_in,
                              void* d_out, int out_size, void* d_ws, size_t ws_size,
                              hipStream_t stream) {
  (void)in_sizes; (void)n_in; (void)out_size; (void)ws_size;
  const float* x    = (const float*)d_in[0];
  const float* srcM = (const float*)d_in[1];
  const float* mask = (const float*)d_in[2];
  const float* Wq   = (const float*)d_in[3];
  const float* bq   = (const float*)d_in[4];
  const float* Wk   = (const float*)d_in[5];
  const float* bk   = (const float*)d_in[6];
  const float* Wv   = (const float*)d_in[7];
  const float* bv   = (const float*)d_in[8];
  const float* Wm   = (const float*)d_in[9];
  const float* bm   = (const float*)d_in[10];
  const float* W1   = (const float*)d_in[11];
  const float* b1   = (const float*)d_in[12];
  const float* W2   = (const float*)d_in[13];
  const float* b2   = (const float*)d_in[14];
  float* out = (float*)d_out;

  char* ws = (char*)d_ws;
  const size_t szYT = (size_t)BATCH * NTOK * 512 * 2;   // concat [x|msg], token-major
  const size_t szST = (size_t)BATCH * NTOK * DMODEL * 2;
  const size_t szHD = (size_t)BATCH * NHEAD * NTOK * DHEAD * 2;
  const size_t szAT = (size_t)BATCH * NTOK * DMODEL * 2;
  const size_t szY1 = (size_t)BATCH * 512 * NTOK * 4;
  const size_t szZT = (size_t)BATCH * NTOK * 512 * 2;

  __bf16* ytok = (__bf16*)(ws);
  __bf16* stok = (__bf16*)(ws + szYT);
  __bf16* qbuf = (__bf16*)(ws + szYT + szST);
  __bf16* kbuf = (__bf16*)(ws + szYT + szST + szHD);
  __bf16* vbuf = (__bf16*)(ws + szYT + szST + 2 * szHD);
  __bf16* atok = (__bf16*)(ws + szYT + szST + 3 * szHD);
  float*  y1f  = (float*) (ws + szYT + szST + 3 * szHD + szAT);
  __bf16* ztok = (__bf16*)(ws + szYT + szST + 3 * szHD + szAT + szY1);
  float2* stat = (float2*)(ws + szYT + szST + 3 * szHD + szAT + szY1 + szZT);

  dim3 gT(NTOK / 32, DMODEL / 32, BATCH);
  transpose_cvt_kernel<<<gT, 256, 0, stream>>>(x, ytok, 512);
  transpose_cvt_kernel<<<gT, 256, 0, stream>>>(srcM, stok, DMODEL);

  const long headStride = (long)NHEAD * NTOK * DHEAD;
  dim3 gP(NTOK / 64, DMODEL / 16, BATCH);
  gemm_bias_kernel<DMODEL, 1><<<gP, 128, 0, stream>>>(
      Wq, bq, ytok, NTOK * 512, 512, qbuf, headStride, 0);
  gemm_bias_kernel<DMODEL, 1><<<gP, 128, 0, stream>>>(
      Wk, bk, stok, NTOK * DMODEL, DMODEL, kbuf, headStride, 0);
  gemm_bias_kernel<DMODEL, 1><<<gP, 128, 0, stream>>>(
      Wv, bv, stok, NTOK * DMODEL, DMODEL, vbuf, headStride, 0);

  dim3 gF(NTOK / 128, NHEAD, BATCH);
  flash_attn_kernel<<<gF, 256, 0, stream>>>(qbuf, kbuf, vbuf, mask, atok);

  // merge conv -> upper half (cols 256..511) of token-major concat buffer
  gemm_bias_kernel<DMODEL, 0><<<gP, 128, 0, stream>>>(
      Wm, bm, atok, NTOK * DMODEL, DMODEL, ytok + DMODEL, NTOK * 512, 512);

  dim3 g1(NTOK / 64, 512 / 16, BATCH);
  gemm_bias_kernel<512, 3><<<g1, 128, 0, stream>>>(
      W1, b1, ytok, NTOK * 512, 512, y1f, 512 * NTOK, 0);

  norm_stats_kernel<<<BATCH * 512, 256, 0, stream>>>(y1f, stat);
  dim3 gA(NTOK / 32, BATCH);
  norm_apply_kernel<<<gA, 256, 0, stream>>>(y1f, stat, ztok);

  dim3 g2(NTOK / 64, DMODEL / 16, BATCH);
  gemm_bias_kernel<512, 3><<<g2, 128, 0, stream>>>(
      W2, b2, ztok, NTOK * 512, 512, out, DMODEL * NTOK, 0);
}
